// DogeDynamicMaskAttention_30399778521228
// MI455X (gfx1250) — compile-verified, matrix operand order checked by reading, not run on hardware
//
#include <hip/hip_runtime.h>
#include <cstdint>
#include <cstddef>

// ---------------- problem constants ----------------
#define BB      2
#define SS      2048
#define HIDDEN  1024
#define NH      8
#define NKV     4
#define DH      128
#define NGROUPS 2          // NH / NKV
#define QKV_N   2048       // NH*DH + 2*NKV*DH
#define ROWS    (BB*SS)    // 4096
static constexpr float ATT_SCALE = 0.08838834764831845f;  // 1/sqrt(128)

// ---------------- types ----------------
typedef __bf16 bf16;
typedef __attribute__((ext_vector_type(16))) __bf16 v16bf;
typedef __attribute__((ext_vector_type(8)))  __bf16 v8bf;
typedef __attribute__((ext_vector_type(8)))  float  v8f;

// Assemble a WMMA 16-bit fragment: per-lane, elements 0..7 are 8 contiguous
// bf16 at p0, elements 8..15 are 8 contiguous bf16 at p1 (two b128 loads).
__device__ __forceinline__ v16bf ld16(const bf16* p0, const bf16* p1) {
  v8bf a = *reinterpret_cast<const v8bf*>(p0);
  v8bf b = *reinterpret_cast<const v8bf*>(p1);
  return __builtin_shufflevector(a, b, 0,1,2,3,4,5,6,7,8,9,10,11,12,13,14,15);
}

__device__ __forceinline__ v8f wmma_bf16(v16bf a, v16bf b, v8f c) {
  // D = A(16x32) * B(32x16) + C, fp32 accumulate
  return __builtin_amdgcn_wmma_f32_16x16x32_bf16(false, a, false, b, (short)0, c,
                                                 false, false);
}

// ---------------- stage 0: casts / transposes ----------------
__global__ void cast_hidden_kernel(const float* __restrict__ x, bf16* __restrict__ y, int n) {
  int i = blockIdx.x * blockDim.x + threadIdx.x;
  if (i < n) y[i] = (bf16)x[i];
}

// Wt_qkv[n][k] : n<1024 -> Wq[k][n]; n<1536 -> Wk[k][n-1024]; else Wv[k][n-1536]
__global__ void build_wqkv_t_kernel(const float* __restrict__ Wq, const float* __restrict__ Wk,
                                    const float* __restrict__ Wv, bf16* __restrict__ Wt) {
  int i = blockIdx.x * blockDim.x + threadIdx.x;      // n*HIDDEN + k
  if (i >= QKV_N * HIDDEN) return;
  int n = i / HIDDEN, k = i % HIDDEN;
  float v;
  if (n < NH * DH)              v = Wq[(size_t)k * (NH * DH) + n];
  else if (n < NH * DH + NKV*DH) v = Wk[(size_t)k * (NKV * DH) + (n - NH * DH)];
  else                           v = Wv[(size_t)k * (NKV * DH) + (n - NH * DH - NKV * DH)];
  Wt[i] = (bf16)v;
}

__global__ void build_wo_t_kernel(const float* __restrict__ Wo, bf16* __restrict__ Wt) {
  int i = blockIdx.x * blockDim.x + threadIdx.x;      // n*1024 + k
  if (i >= HIDDEN * HIDDEN) return;
  int n = i / HIDDEN, k = i % HIDDEN;
  Wt[i] = (bf16)Wo[(size_t)k * HIDDEN + n];
}

// ---------------- generic bf16 WMMA GEMM: C[M,N] = A[M,K] * Bt[N,K]^T -------
// one wave computes a 16x64 tile (A-fragment reused across 4 B tiles)
__global__ __launch_bounds__(32) void gemm_bf16_kernel(const bf16* __restrict__ A,
                                                       const bf16* __restrict__ Bt,
                                                       float* __restrict__ C,
                                                       int M, int N, int K, int store_bf16,
                                                       bf16* __restrict__ Cb) {
  const int lane = threadIdx.x;
  const int half = lane >> 4;
  const int ln   = lane & 15;
  const int kb   = half ? 8 : 0;
  const int mt   = blockIdx.x * 16;
  const int nt   = blockIdx.y * 64;

  v8f acc[4];
#pragma unroll
  for (int i = 0; i < 4; i++)
#pragma unroll
    for (int r = 0; r < 8; r++) acc[i][r] = 0.0f;

  for (int kc = 0; kc < K; kc += 32) {
    const bf16* ab = A + (size_t)(mt + ln) * K + kc;
    v16bf af = ld16(ab + kb, ab + 16 + kb);
#pragma unroll
    for (int nb = 0; nb < 4; nb++) {
      const bf16* bb = Bt + (size_t)(nt + nb * 16 + ln) * K + kc;
      v16bf bfr = ld16(bb + kb, bb + 16 + kb);
      acc[nb] = wmma_bf16(af, bfr, acc[nb]);
    }
  }
  // C layout: slot r = row (r + 8*half), column = lane&15 within each 16-tile
#pragma unroll
  for (int nb = 0; nb < 4; nb++)
#pragma unroll
    for (int r = 0; r < 8; r++) {
      size_t idx = (size_t)(mt + r + 8 * half) * N + nt + nb * 16 + ln;
      if (store_bf16) Cb[idx] = (bf16)acc[nb][r];
      else            C[idx]  = acc[nb][r];
    }
}

// ---------------- stage 2: RoPE + reorder into attention layouts ------------
// qkv fp32 [ROWS][2048] ->
//   q  bf16 [B][H][S][D]   (roped)
//   k  bf16 [B][KV][S][D]  (roped)
//   vt bf16 [B][KV][D][S]  (transposed V, for PV B-fragments)
__global__ void rope_reorder_kernel(const float* __restrict__ qkv,
                                    const float* __restrict__ cosT,
                                    const float* __restrict__ sinT,
                                    bf16* __restrict__ q, bf16* __restrict__ k,
                                    bf16* __restrict__ vt) {
  size_t i = (size_t)blockIdx.x * blockDim.x + threadIdx.x;
  if (i >= (size_t)ROWS * QKV_N) return;
  int row = (int)(i / QKV_N);
  int col = (int)(i % QKV_N);
  int b = row / SS, s = row % SS;
  float x = qkv[i];
  if (col < NH * DH + NKV * DH) {   // q or k : apply RoPE
    int d = col & (DH - 1);
    float c  = cosT[(size_t)s * DH + d];
    float sn = sinT[(size_t)s * DH + d];
    float other = (d < DH / 2) ? -qkv[(size_t)row * QKV_N + col + DH / 2]
                               :  qkv[(size_t)row * QKV_N + col - DH / 2];
    bf16 bv = (bf16)(x * c + other * sn);
    if (col < NH * DH) {
      int h = col >> 7;
      q[(((size_t)(b * NH + h)) * SS + s) * DH + d] = bv;
    } else {
      int kvh = (col - NH * DH) >> 7;
      k[(((size_t)(b * NKV + kvh)) * SS + s) * DH + d] = bv;
    }
  } else {                          // v -> transposed layout [kv][d][S]
    int cc = col - NH * DH - NKV * DH;
    int kvh = cc >> 7, d = cc & (DH - 1);
    vt[(((size_t)(b * NKV + kvh)) * DH + d) * SS + s] = (bf16)x;
  }
}

// ---------------- stage 3: dynamic bias  dyn[b][h][s] ----------------
__global__ void dyn_kernel(const float* __restrict__ qkv, const float* __restrict__ Wdt,
                           const float* __restrict__ Avec, float* __restrict__ dyn) {
  int idx = blockIdx.x * blockDim.x + threadIdx.x;   // (b*S+s)*H + h
  if (idx >= ROWS * NH) return;
  int h = idx % NH, rs = idx / NH;
  const float* vrow = qkv + (size_t)rs * QKV_N + NH * DH + NKV * DH;   // raw V (pre-RoPE)
  float acc = 0.0f;
  for (int j = 0; j < NKV * DH; j++) acc += vrow[j] * Wdt[(size_t)j * NH + h];
  float sp = (acc > 20.0f) ? acc : log1pf(__expf(acc));                 // softplus
  int b = rs / SS, s = rs % SS;
  dyn[((size_t)(b * NH + h)) * SS + s] = __expf(Avec[h] * sp);
}

// ---------------- stage 4: flash attention (1 wave = 16 query rows) ---------
__global__ __launch_bounds__(32) void attn_kernel(const bf16* __restrict__ q,
                                                  const bf16* __restrict__ k,
                                                  const bf16* __restrict__ vt,
                                                  const float* __restrict__ dyn,
                                                  bf16* __restrict__ out) {
  const int qt   = blockIdx.x;           // query tile (16 rows)
  const int h    = blockIdx.y;
  const int b    = blockIdx.z;
  const int kvh  = h / NGROUPS;
  const int lane = threadIdx.x;
  const int half = lane >> 4;
  const int ln   = lane & 15;
  const int kb   = half ? 8 : 0;
  const int qbase = qt * 16;

  __shared__ __align__(16) bf16 Pl[16][40];   // 80B rows -> 16B aligned slices

  const bf16*  qp   = q   + ((size_t)(b * NH  + h  ) * SS + qbase) * DH;
  const bf16*  kp   = k   + ((size_t)(b * NKV + kvh) * SS) * DH;
  const bf16*  vtp  = vt  + ((size_t)(b * NKV + kvh) * DH) * SS;
  const float* dynp = dyn + ((size_t)(b * NH  + h  ) * SS);

  // Q A-fragments for the 4 K-chunks of D=128 (kept in registers all loop)
  v16bf qa[4];
#pragma unroll
  for (int dc = 0; dc < 4; dc++) {
    const bf16* base = qp + (size_t)ln * DH + dc * 32;
    qa[dc] = ld16(base + kb, base + 16 + kb);
  }

  float mrow[8], lrow[8];
  v8f o[8];
#pragma unroll
  for (int r = 0; r < 8; r++) { mrow[r] = -1e30f; lrow[r] = 0.0f; }
#pragma unroll
  for (int nb = 0; nb < 8; nb++)
#pragma unroll
    for (int r = 0; r < 8; r++) o[nb][r] = 0.0f;

  const int nchunks = (qbase + 16 + 31) / 32;
  for (int c = 0; c < nchunks; c++) {
    const int kc = c * 32;
    // ---- scores: two 16x16 tiles over 32 keys, K=128 contraction ----
    v8f sc[2];
#pragma unroll
    for (int t = 0; t < 2; t++) {
#pragma unroll
      for (int r = 0; r < 8; r++) sc[t][r] = 0.0f;
#pragma unroll
      for (int dc = 0; dc < 4; dc++) {
        const bf16* base = kp + (size_t)(kc + t * 16 + ln) * DH + dc * 32;
        v16bf kf = ld16(base + kb, base + 16 + kb);
        sc[t] = wmma_bf16(qa[dc], kf, sc[t]);
      }
    }
    // ---- bias + causal mask + online softmax ----
    float sv[2][8], tmax[8];
#pragma unroll
    for (int r = 0; r < 8; r++) tmax[r] = -1e30f;
#pragma unroll
    for (int t = 0; t < 2; t++) {
      const int key = kc + t * 16 + ln;
      const float bias = dynp[key];
#pragma unroll
      for (int r = 0; r < 8; r++) {
        const int qrow = qbase + r + 8 * half;
        float s = sc[t][r] * ATT_SCALE + bias;
        s = (key > qrow) ? -1e30f : s;
        sv[t][r] = s;
        tmax[r] = fmaxf(tmax[r], s);
      }
    }
#pragma unroll
    for (int mask = 1; mask < 16; mask <<= 1)
#pragma unroll
      for (int r = 0; r < 8; r++) tmax[r] = fmaxf(tmax[r], __shfl_xor(tmax[r], mask, 32));
    float scl[8], rsum[8];
#pragma unroll
    for (int r = 0; r < 8; r++) {
      float mn = fmaxf(mrow[r], tmax[r]);
      scl[r] = __expf(mrow[r] - mn);
      mrow[r] = mn;
      rsum[r] = 0.0f;
    }
#pragma unroll
    for (int t = 0; t < 2; t++)
#pragma unroll
      for (int r = 0; r < 8; r++) {
        float p = __expf(sv[t][r] - mrow[r]);   // masked (-1e30) -> 0
        rsum[r] += p;
        Pl[r + 8 * half][t * 16 + ln] = (bf16)p;
      }
#pragma unroll
    for (int mask = 1; mask < 16; mask <<= 1)
#pragma unroll
      for (int r = 0; r < 8; r++) rsum[r] += __shfl_xor(rsum[r], mask, 32);
#pragma unroll
    for (int r = 0; r < 8; r++) lrow[r] = lrow[r] * scl[r] + rsum[r];
#pragma unroll
    for (int nb = 0; nb < 8; nb++)
#pragma unroll
      for (int r = 0; r < 8; r++) o[nb][r] *= scl[r];
    __syncthreads();
    // P as A-fragment (16x32 over this chunk's keys), read back in A layout
    v16bf pa = ld16(&Pl[ln][kb], &Pl[ln][16 + kb]);
    __syncthreads();
    // ---- O += P * V  (V^T rows are contiguous over keys) ----
#pragma unroll
    for (int nb = 0; nb < 8; nb++) {
      const bf16* base = vtp + (size_t)(nb * 16 + ln) * SS + kc;
      v16bf vf = ld16(base + kb, base + 16 + kb);
      o[nb] = wmma_bf16(pa, vf, o[nb]);
    }
  }
  // ---- normalize + store bf16 [B*S][H*D] for the Wo GEMM ----
  float inv[8];
#pragma unroll
  for (int r = 0; r < 8; r++) inv[r] = 1.0f / lrow[r];
#pragma unroll
  for (int nb = 0; nb < 8; nb++)
#pragma unroll
    for (int r = 0; r < 8; r++) {
      const int m = r + 8 * half;
      out[(size_t)(b * SS + qbase + m) * (NH * DH) + h * DH + nb * 16 + ln] =
          (bf16)(o[nb][r] * inv[r]);
    }
}

// ---------------- host-side orchestration ----------------
extern "C" void kernel_launch(void* const* d_in, const int* in_sizes, int n_in,
                              void* d_out, int out_size, void* d_ws, size_t ws_size,
                              hipStream_t stream) {
  const float* hidden = (const float*)d_in[0];
  const float* Wq     = (const float*)d_in[1];
  const float* Wk     = (const float*)d_in[2];
  const float* Wv     = (const float*)d_in[3];
  const float* Wdt    = (const float*)d_in[4];
  const float* Avec   = (const float*)d_in[5];
  const float* Wo     = (const float*)d_in[6];
  const float* cosT   = (const float*)d_in[7];
  const float* sinT   = (const float*)d_in[8];
  // d_in[9] = causal_mask : causality computed analytically

  // workspace carve-up (all 256B aligned)
  char* ws = (char*)d_ws;
  size_t off = 0;
  auto alloc = [&](size_t bytes) { void* p = ws + off; off += (bytes + 255) & ~(size_t)255; return p; };
  bf16*  hid_b   = (bf16*) alloc((size_t)ROWS * HIDDEN * 2);      // 8 MiB
  bf16*  wt_qkv  = (bf16*) alloc((size_t)QKV_N * HIDDEN * 2);     // 4 MiB
  bf16*  wt_o    = (bf16*) alloc((size_t)HIDDEN * HIDDEN * 2);    // 2 MiB
  float* qkv_f   = (float*)alloc((size_t)ROWS * QKV_N * 4);       // 32 MiB
  bf16*  qbuf    = (bf16*) alloc((size_t)BB * NH  * SS * DH * 2); // 8 MiB
  bf16*  kbuf    = (bf16*) alloc((size_t)BB * NKV * SS * DH * 2); // 4 MiB
  bf16*  vtbuf   = (bf16*) alloc((size_t)BB * NKV * DH * SS * 2); // 4 MiB
  float* dynbuf  = (float*)alloc((size_t)BB * NH * SS * 4);       // 128 KiB
  bf16*  attn_o  = (bf16*) alloc((size_t)ROWS * HIDDEN * 2);      // 8 MiB
  (void)ws_size;

  const int T = 256;
  // stage 0: casts / transposes
  {
    int n = ROWS * HIDDEN;
    cast_hidden_kernel<<<(n + T - 1) / T, T, 0, stream>>>(hidden, hid_b, n);
    int n2 = QKV_N * HIDDEN;
    build_wqkv_t_kernel<<<(n2 + T - 1) / T, T, 0, stream>>>(Wq, Wk, Wv, wt_qkv);
    int n3 = HIDDEN * HIDDEN;
    build_wo_t_kernel<<<(n3 + T - 1) / T, T, 0, stream>>>(Wo, wt_o);
  }
  // stage 1: QKV projection (WMMA) -> fp32
  gemm_bf16_kernel<<<dim3(ROWS / 16, QKV_N / 64), 32, 0, stream>>>(
      hid_b, wt_qkv, qkv_f, ROWS, QKV_N, HIDDEN, 0, nullptr);
  // stage 2: RoPE + layout
  {
    size_t n = (size_t)ROWS * QKV_N;
    rope_reorder_kernel<<<(unsigned)((n + T - 1) / T), T, 0, stream>>>(
        qkv_f, cosT, sinT, qbuf, kbuf, vtbuf);
  }
  // stage 3: dynamic bias
  {
    int n = ROWS * NH;
    dyn_kernel<<<(n + T - 1) / T, T, 0, stream>>>(qkv_f, Wdt, Avec, dynbuf);
  }
  // stage 4: flash attention (WMMA)
  attn_kernel<<<dim3(SS / 16, NH, BB), 32, 0, stream>>>(qbuf, kbuf, vtbuf, dynbuf, attn_o);
  // stage 5: output projection (WMMA) -> fp32 d_out
  gemm_bf16_kernel<<<dim3(ROWS / 16, HIDDEN / 64), 32, 0, stream>>>(
      attn_o, wt_o, (float*)d_out, ROWS, HIDDEN, HIDDEN, 0, nullptr);
  (void)in_sizes; (void)n_in; (void)out_size;
}
